// FactorizedAttention_75582834475582
// MI455X (gfx1250) — matrix operand order repaired and checked by reading, compiled, not run on hardware
//
#include <hip/hip_runtime.h>
#include <hip/hip_bf16.h>

// ---------------- CDNA5 WMMA types/helpers (wave32) ----------------
typedef __attribute__((ext_vector_type(16))) __bf16 v16bf;
typedef __attribute__((ext_vector_type(4)))  __bf16 v4bf;
typedef __attribute__((ext_vector_type(2)))  __bf16 v2bf;
typedef __attribute__((ext_vector_type(8)))  float  v8f;
typedef __attribute__((ext_vector_type(4)))  float  v4f;
typedef __attribute__((ext_vector_type(2)))  float  v2f;

struct U8x32 { uint4 lo; uint4 hi; };   // 32 bytes, 16B aligned

// packed converts: lower to v_cvt_pk_bf16_f32 (vector convert, no bit-merge chains)
__device__ __forceinline__ unsigned int pk2(float a, float b) {
    v2f f; f[0] = a; f[1] = b;
    v2bf h = __builtin_convertvector(f, v2bf);
    return __builtin_bit_cast(unsigned int, h);
}
__device__ __forceinline__ uint2 pk4(float a, float b, float c, float d) {
    v4f f; f[0] = a; f[1] = b; f[2] = c; f[3] = d;
    v4bf h = __builtin_convertvector(f, v4bf);
    return __builtin_bit_cast(uint2, h);
}

// A fragment from row-major LDS tile (stride in elems, rows 16B aligned):
// lane holds row m = lane&15; elements = K kh..kh+15 (kh = 16*(lane>=16)).
__device__ __forceinline__ v16bf frag_a_row(const __bf16* As, int strideElems, int mbase, int lane) {
    const int m  = mbase + (lane & 15);
    const int kh = (lane >> 4) << 4;
    const uint4* p = (const uint4*)((const char*)As + (size_t)m * strideElems * 2 + (size_t)kh * 2);
    U8x32 u; u.lo = p[0]; u.hi = p[1];   // +0B and +16B
    return __builtin_bit_cast(v16bf, u);
}

// Fragment-ordered B storage: group g = n*kgroups + kq holds B[kq*16+0..15][n] contiguously.
__device__ __forceinline__ v16bf frag_b_pk(const __bf16* Bf, int kgroups, int kqbase, int nbase, int lane) {
    const int grp = (nbase + (lane & 15)) * kgroups + kqbase + (lane >> 4);
    const uint4* p = (const uint4*)((const char*)Bf + (size_t)grp * 32);
    U8x32 u; u.lo = p[0]; u.hi = p[1];
    return __builtin_bit_cast(v16bf, u);
}

// stage B[k..k+3][n] (k = rg*4) as one packed b64 store into fragment-ordered LDS
__device__ __forceinline__ void stage_b4(__bf16* Bf, int kgroups, int n, int rg,
                                         float x0, float x1, float x2, float x3) {
    const int grp = n * kgroups + (rg >> 2);
    *(uint2*)((char*)Bf + (size_t)grp * 32 + (size_t)(rg & 3) * 8) = pk4(x0, x1, x2, x3);
}

__device__ __forceinline__ v8f wmma_bf16(v16bf a, v16bf b, v8f c) {
    return __builtin_amdgcn_wmma_f32_16x16x32_bf16(false, a, false, b, (short)0, c, false, false);
}

// ---------------- problem constants ----------------
#define BATCH    32
#define SEQ      1025
#define DIM_     512
#define HEADS_   8
#define HD       64
#define M_TOTAL  (BATCH * SEQ)  // 32800
#define IMG      32

#define BM 128
#define BN 128
#define BKK 32
#define SA 40          // A-tile row stride in bf16 elems (80B, 16B aligned)

// ============ Kernel 1: QKV GEMM (32800x512 @ 512x1536) -> head-major q,k,v ============
__global__ __launch_bounds__(256) void qkv_gemm_kernel(
    const float* __restrict__ X, const float* __restrict__ Wqkv,
    const float* __restrict__ bqkv,
    float* __restrict__ qb, float* __restrict__ kb, float* __restrict__ vb)
{
    __shared__ alignas(16) __bf16 As[BM * SA];
    __shared__ alignas(16) __bf16 Bf[BN * 2 * 16];

    const int tid  = threadIdx.x;
    const int lane = tid & 31;
    const int wid  = tid >> 5;
    const int wm   = (wid & 3) * 32;
    const int wn   = (wid >> 2) * 64;
    const int m0   = blockIdx.x * BM;
    const int n0   = blockIdx.y * BN;

    float4 aReg[4];
    float  bReg[4][4];
    bool   aOK[4];

    auto load_tiles = [&](int k0) {
#pragma unroll
        for (int i = 0; i < 4; ++i) {
            const int idx = i * 256 + tid;
            const int r = idx >> 3, cq = idx & 7;
            const int gm  = m0 + r;
            const int gmc = (gm < M_TOTAL) ? gm : (M_TOTAL - 1);   // clamp: always valid
            aOK[i]  = gm < M_TOTAL;
            aReg[i] = *(const float4*)&X[(size_t)gmc * DIM_ + k0 + cq * 4];
        }
#pragma unroll
        for (int i = 0; i < 4; ++i) {
            const int idx = i * 256 + tid;
            const int n = idx & 127, rg = idx >> 7;
            const float* col = &Wqkv[(size_t)(k0 + rg * 4) * 1536 + n0 + n];
            bReg[i][0] = col[0]; bReg[i][1] = col[1536];
            bReg[i][2] = col[2 * 1536]; bReg[i][3] = col[3 * 1536];
        }
    };
    auto store_tiles = [&]() {
#pragma unroll
        for (int i = 0; i < 4; ++i) {
            const int idx = i * 256 + tid;
            const int r = idx >> 3, cq = idx & 7;
            uint2 u = pk4(aReg[i].x, aReg[i].y, aReg[i].z, aReg[i].w);
            if (!aOK[i]) { u.x = 0u; u.y = 0u; }
            *(uint2*)((char*)As + (size_t)r * SA * 2 + (size_t)cq * 8) = u;
        }
#pragma unroll
        for (int i = 0; i < 4; ++i) {
            const int idx = i * 256 + tid;
            const int n = idx & 127, rg = idx >> 7;
            stage_b4(Bf, 2, n, rg, bReg[i][0], bReg[i][1], bReg[i][2], bReg[i][3]);
        }
    };

    v8f acc[2][4] = {};
    load_tiles(0);
    for (int k0 = 0; k0 < DIM_; k0 += BKK) {
        store_tiles();
        __syncthreads();
        if (k0 + BKK < DIM_) load_tiles(k0 + BKK);   // overlap next-tile loads with WMMA

        v16bf af[2], bf[4];
#pragma unroll
        for (int i = 0; i < 2; ++i) af[i] = frag_a_row(As, SA, wm + 16 * i, lane);
#pragma unroll
        for (int j = 0; j < 4; ++j) bf[j] = frag_b_pk(Bf, 2, 0, wn + 16 * j, lane);
#pragma unroll
        for (int i = 0; i < 2; ++i)
#pragma unroll
            for (int j = 0; j < 4; ++j)
                acc[i][j] = wmma_bf16(af[i], bf[j], acc[i][j]);
        __syncthreads();
    }

    // epilogue: bias + scatter into q/k/v [b][h][n][d]
#pragma unroll
    for (int i = 0; i < 2; ++i) {
#pragma unroll
        for (int j = 0; j < 4; ++j) {
#pragma unroll
            for (int e = 0; e < 8; ++e) {
                const int gm = m0 + wm + i * 16 + e + ((lane >> 4) << 3);
                const int gn = n0 + wn + j * 16 + (lane & 15);
                if (gm < M_TOTAL) {
                    const float v = acc[i][j][e] + bqkv[gn];
                    const int s  = gn >> 9;
                    const int dd = gn & 511;
                    const int h  = dd >> 6;
                    const int d  = dd & 63;
                    const int bi = gm / SEQ;
                    const int ni = gm - bi * SEQ;
                    float* dst = (s == 0) ? qb : (s == 1) ? kb : vb;
                    dst[(((size_t)bi * HEADS_ + h) * SEQ + ni) * HD + d] = v;
                }
            }
        }
    }
}

// ============ Kernel 2: per-(b,h,d) softmax stats over n ============
__global__ __launch_bounds__(256) void ksoftmax_stats_kernel(
    const float* __restrict__ kbuf, float* __restrict__ kmax, float* __restrict__ ksum)
{
    const int col = blockIdx.x * 256 + threadIdx.x;
    const int bh = col >> 6, d = col & 63;
    const float* kp = kbuf + ((size_t)bh * SEQ) * HD + d;
    float m = -3.0e38f;
    for (int n = 0; n < SEQ; ++n) m = fmaxf(m, kp[(size_t)n * HD]);
    float s = 0.0f;
    for (int n = 0; n < SEQ; ++n) s += __expf(kp[(size_t)n * HD] - m);
    kmax[col] = m;
    ksum[col] = s;
}

// ============ Kernel 3: context[bh] = softmax(k)^T @ v  (64x64, K=1025) ============
__global__ __launch_bounds__(256) void context_kernel(
    const float* __restrict__ kbuf, const float* __restrict__ vbuf,
    const float* __restrict__ kmax, const float* __restrict__ ksum,
    float* __restrict__ ctx)
{
    __shared__ alignas(16) __bf16 Kt[64 * SA];       // transposed softmaxed k: Kt[p][n_local]
    __shared__ alignas(16) __bf16 Vf[64 * 2 * 16];   // fragment-ordered v
    const int bh  = blockIdx.x;
    const int tid = threadIdx.x, lane = tid & 31, wid = tid >> 5;
    const float* kp = kbuf + (size_t)bh * SEQ * HD;
    const float* vp = vbuf + (size_t)bh * SEQ * HD;

    const int dstage = tid & 63;
    const float km   = kmax[bh * 64 + dstage];
    const float kinv = 1.0f / ksum[bh * 64 + dstage];

    float kReg[2][4], vReg[2][4];
    auto ctx_load = [&](int n0) {
#pragma unroll
        for (int i = 0; i < 2; ++i) {
            const int idx = i * 256 + tid;
            const int d = idx & 63, ng = idx >> 6;
            const int nl = ng * 4;
            const float* kq = kp + (size_t)(n0 + nl) * HD + d;
            const float* vq = vp + (size_t)(n0 + nl) * HD + d;
            kReg[i][0] = kq[0]; kReg[i][1] = kq[HD]; kReg[i][2] = kq[2 * HD]; kReg[i][3] = kq[3 * HD];
            vReg[i][0] = vq[0]; vReg[i][1] = vq[HD]; vReg[i][2] = vq[2 * HD]; vReg[i][3] = vq[3 * HD];
        }
    };
    auto ctx_store = [&]() {
#pragma unroll
        for (int i = 0; i < 2; ++i) {
            const int idx = i * 256 + tid;
            const int d = idx & 63, ng = idx >> 6;
            const int nl = ng * 4;
            const float e0 = __expf(kReg[i][0] - km) * kinv;
            const float e1 = __expf(kReg[i][1] - km) * kinv;
            const float e2 = __expf(kReg[i][2] - km) * kinv;
            const float e3 = __expf(kReg[i][3] - km) * kinv;
            *(uint2*)((char*)Kt + (size_t)d * SA * 2 + (size_t)nl * 2) = pk4(e0, e1, e2, e3);
            stage_b4(Vf, 2, d, ng, vReg[i][0], vReg[i][1], vReg[i][2], vReg[i][3]);
        }
    };

    v8f acc[2] = {};
    ctx_load(0);
    for (int n0 = 0; n0 < 1024; n0 += 32) {          // 32 full tiles, pipelined
        ctx_store();
        __syncthreads();
        if (n0 + 32 < 1024) ctx_load(n0 + 32);
#pragma unroll
        for (int s = 0; s < 2; ++s) {
            const int sub = wid * 2 + s;
            const int psub = sub >> 2, qsub = sub & 3;
            v16bf a = frag_a_row(Kt, SA, psub * 16, lane);
            v16bf b = frag_b_pk(Vf, 2, 0, qsub * 16, lane);
            acc[s] = wmma_bf16(a, b, acc[s]);
        }
        __syncthreads();
    }
    // tail tile: only n = 1024 valid
    {
#pragma unroll
        for (int i = 0; i < 2; ++i) {
            const int idx = i * 256 + tid;
            const int d = idx & 63, ng = idx >> 6;
            const int nl = ng * 4;
            float e0 = 0.0f, v0 = 0.0f;
            if (nl == 0) {
                e0 = __expf(kp[(size_t)1024 * HD + d] - km) * kinv;
                v0 = vp[(size_t)1024 * HD + d];
            }
            *(uint2*)((char*)Kt + (size_t)d * SA * 2 + (size_t)nl * 2) = pk4(e0, 0.0f, 0.0f, 0.0f);
            stage_b4(Vf, 2, d, ng, v0, 0.0f, 0.0f, 0.0f);
        }
        __syncthreads();
#pragma unroll
        for (int s = 0; s < 2; ++s) {
            const int sub = wid * 2 + s;
            const int psub = sub >> 2, qsub = sub & 3;
            v16bf a = frag_a_row(Kt, SA, psub * 16, lane);
            v16bf b = frag_b_pk(Vf, 2, 0, qsub * 16, lane);
            acc[s] = wmma_bf16(a, b, acc[s]);
        }
    }
#pragma unroll
    for (int s = 0; s < 2; ++s) {
        const int sub = wid * 2 + s;
        const int psub = sub >> 2, qsub = sub & 3;
#pragma unroll
        for (int e = 0; e < 8; ++e) {
            const int p = psub * 16 + e + ((lane >> 4) << 3);
            const int q = qsub * 16 + (lane & 15);
            ctx[((size_t)bh * 64 + p) * 64 + q] = acc[s][e];
        }
    }
}

// ============ Kernel 4: depthwise convs (3/5/7) on v image -> ev ============
__global__ __launch_bounds__(256) void crpe_conv_kernel(
    const float* __restrict__ vbuf,
    const float* __restrict__ w3, const float* __restrict__ b3,
    const float* __restrict__ w5, const float* __restrict__ b5,
    const float* __restrict__ w7, const float* __restrict__ b7,
    float* __restrict__ ev)
{
    const long long t = (long long)blockIdx.x * 256 + threadIdx.x;
    const int d = (int)(t & 63);
    const int x = (int)((t >> 6) & 31);
    const int y = (int)((t >> 11) & 31);
    const int g = (int)(t >> 16);
    const int h = g & 7, b = g >> 3;
    const int c = h * HD + d;

    const float* w; float bias; int ks;
    if (c < 128)      { w = w3 + (size_t)c * 9;          bias = b3[c];       ks = 3; }
    else if (c < 320) { w = w5 + (size_t)(c - 128) * 25; bias = b5[c - 128]; ks = 5; }
    else              { w = w7 + (size_t)(c - 320) * 49; bias = b7[c - 320]; ks = 7; }
    const int p = ks >> 1;

    const float* vp = vbuf + (((size_t)(b * HEADS_ + h)) * SEQ + 1) * HD + d;
    float acc = bias;
    for (int ky = 0; ky < ks; ++ky) {
        const int yy = y + ky - p;
        if (yy < 0 || yy >= IMG) continue;
        for (int kx = 0; kx < ks; ++kx) {
            const int xx = x + kx - p;
            if (xx < 0 || xx >= IMG) continue;
            acc = fmaf(w[ky * ks + kx], vp[(size_t)(yy * IMG + xx) * HD], acc);
        }
    }
    ev[(((size_t)(b * HEADS_ + h)) * SEQ + 1 + y * IMG + x) * HD + d] = acc;
}

// ============ Kernel 5: factor = (q*scale) @ context, fused crpe = q*ev ============
__global__ __launch_bounds__(256) void factor_crpe_kernel(
    const float* __restrict__ qbuf, const float* __restrict__ ctx,
    const float* __restrict__ ev, float* __restrict__ attn)
{
    __shared__ alignas(16) __bf16 Qs[128 * SA];
    __shared__ alignas(16) __bf16 Cf[64 * 4 * 16];   // fragment-ordered context (K=64)
    const int bh = blockIdx.x;
    const int mt = blockIdx.y;
    const int tid = threadIdx.x, lane = tid & 31, wid = tid >> 5;
    const float* qp = qbuf + (size_t)bh * SEQ * HD;
    const int m0 = mt * 128;

#pragma unroll
    for (int i = 0; i < 4; ++i) {
        const int idx = i * 256 + tid;
        const int q = idx & 63, rg = idx >> 6;
        const float* col = &ctx[((size_t)bh * 64 + rg * 4) * 64 + q];
        stage_b4(Cf, 4, q, rg, col[0], col[64], col[128], col[192]);
    }
    // stage q tile 128x64 with clamped loads + select-zero (no divergent loads)
#pragma unroll
    for (int i = 0; i < 8; ++i) {
        const int idx = i * 256 + tid;
        const int r = idx >> 4, cq = idx & 15;
        const int n  = m0 + r;
        const int nc = (n < SEQ) ? n : (SEQ - 1);
        const bool ok = n < SEQ;
        const float4 f = *(const float4*)&qp[(size_t)nc * HD + cq * 4];
        uint2 u = pk4(f.x, f.y, f.z, f.w);
        if (!ok) { u.x = 0u; u.y = 0u; }
        *(uint2*)((char*)Qs + (size_t)r * SA * 2 + (size_t)cq * 8) = u;
    }
    __syncthreads();

    v8f acc[4] = {};
#pragma unroll
    for (int kq = 0; kq < 4; kq += 2) {
        v16bf a = frag_a_row(Qs + kq * 16, SA, wid * 16, lane);
#pragma unroll
        for (int j = 0; j < 4; ++j) {
            v16bf b = frag_b_pk(Cf, 4, kq, j * 16, lane);
            acc[j] = wmma_bf16(a, b, acc[j]);
        }
    }

    const float scale = 0.125f;
    const int bi = bh >> 3, hh = bh & 7;
#pragma unroll
    for (int j = 0; j < 4; ++j) {
#pragma unroll
        for (int e = 0; e < 8; ++e) {
            const int n = m0 + wid * 16 + e + ((lane >> 4) << 3);
            const int dcol = j * 16 + (lane & 15);
            if (n < SEQ) {
                const float qv  = qp[(size_t)n * HD + dcol];
                const float evv = (n > 0) ? ev[((size_t)bh * SEQ + n) * HD + dcol] : 0.0f;
                attn[((size_t)(bi * SEQ + n)) * DIM_ + hh * HD + dcol] =
                    acc[j][e] * scale + qv * evv;
            }
        }
    }
}

// ============ Kernel 6: out = attn(32800x512) @ W_out(512x512) + b_out ============
__global__ __launch_bounds__(256) void out_gemm_kernel(
    const float* __restrict__ A, const float* __restrict__ Wout,
    const float* __restrict__ bout, float* __restrict__ out)
{
    __shared__ alignas(16) __bf16 As[BM * SA];
    __shared__ alignas(16) __bf16 Bf[BN * 2 * 16];

    const int tid  = threadIdx.x;
    const int lane = tid & 31;
    const int wid  = tid >> 5;
    const int wm   = (wid & 3) * 32;
    const int wn   = (wid >> 2) * 64;
    const int m0   = blockIdx.x * BM;
    const int n0   = blockIdx.y * BN;

    float4 aReg[4];
    float  bReg[4][4];
    bool   aOK[4];

    auto load_tiles = [&](int k0) {
#pragma unroll
        for (int i = 0; i < 4; ++i) {
            const int idx = i * 256 + tid;
            const int r = idx >> 3, cq = idx & 7;
            const int gm  = m0 + r;
            const int gmc = (gm < M_TOTAL) ? gm : (M_TOTAL - 1);
            aOK[i]  = gm < M_TOTAL;
            aReg[i] = *(const float4*)&A[(size_t)gmc * DIM_ + k0 + cq * 4];
        }
#pragma unroll
        for (int i = 0; i < 4; ++i) {
            const int idx = i * 256 + tid;
            const int n = idx & 127, rg = idx >> 7;
            const float* col = &Wout[(size_t)(k0 + rg * 4) * DIM_ + n0 + n];
            bReg[i][0] = col[0]; bReg[i][1] = col[DIM_];
            bReg[i][2] = col[2 * DIM_]; bReg[i][3] = col[3 * DIM_];
        }
    };
    auto store_tiles = [&]() {
#pragma unroll
        for (int i = 0; i < 4; ++i) {
            const int idx = i * 256 + tid;
            const int r = idx >> 3, cq = idx & 7;
            uint2 u = pk4(aReg[i].x, aReg[i].y, aReg[i].z, aReg[i].w);
            if (!aOK[i]) { u.x = 0u; u.y = 0u; }
            *(uint2*)((char*)As + (size_t)r * SA * 2 + (size_t)cq * 8) = u;
        }
#pragma unroll
        for (int i = 0; i < 4; ++i) {
            const int idx = i * 256 + tid;
            const int n = idx & 127, rg = idx >> 7;
            stage_b4(Bf, 2, n, rg, bReg[i][0], bReg[i][1], bReg[i][2], bReg[i][3]);
        }
    };

    v8f acc[2][4] = {};
    load_tiles(0);
    for (int k0 = 0; k0 < DIM_; k0 += BKK) {
        store_tiles();
        __syncthreads();
        if (k0 + BKK < DIM_) load_tiles(k0 + BKK);

        v16bf af[2], bf[4];
#pragma unroll
        for (int i = 0; i < 2; ++i) af[i] = frag_a_row(As, SA, wm + 16 * i, lane);
#pragma unroll
        for (int j = 0; j < 4; ++j) bf[j] = frag_b_pk(Bf, 2, 0, wn + 16 * j, lane);
#pragma unroll
        for (int i = 0; i < 2; ++i)
#pragma unroll
            for (int j = 0; j < 4; ++j)
                acc[i][j] = wmma_bf16(af[i], bf[j], acc[i][j]);
        __syncthreads();
    }

#pragma unroll
    for (int i = 0; i < 2; ++i) {
#pragma unroll
        for (int j = 0; j < 4; ++j) {
#pragma unroll
            for (int e = 0; e < 8; ++e) {
                const int gm = m0 + wm + i * 16 + e + ((lane >> 4) << 3);
                const int gn = n0 + wn + j * 16 + (lane & 15);
                if (gm < M_TOTAL)
                    out[(size_t)gm * DIM_ + gn] = acc[i][j][e] + bout[gn];
            }
        }
    }
}

// ---------------- launcher ----------------
extern "C" void kernel_launch(void* const* d_in, const int* in_sizes, int n_in,
                              void* d_out, int out_size, void* d_ws, size_t ws_size,
                              hipStream_t stream) {
    const float* x    = (const float*)d_in[0];
    const float* Wqkv = (const float*)d_in[1];
    const float* bqkv = (const float*)d_in[2];
    const float* w3   = (const float*)d_in[3];
    const float* b3   = (const float*)d_in[4];
    const float* w5   = (const float*)d_in[5];
    const float* b5   = (const float*)d_in[6];
    const float* w7   = (const float*)d_in[7];
    const float* b7   = (const float*)d_in[8];
    const float* Wout = (const float*)d_in[9];
    const float* bout = (const float*)d_in[10];

    float* ws = (float*)d_ws;
    const size_t SQ = (size_t)BATCH * HEADS_ * SEQ * HD;
    float* qb   = ws;
    float* kb   = qb + SQ;
    float* vb   = kb + SQ;
    float* kmax = vb + SQ;
    float* ksum = kmax + (size_t)BATCH * HEADS_ * HD;
    float* ctx  = ksum + (size_t)BATCH * HEADS_ * HD;
    float* ev   = ctx + (size_t)BATCH * HEADS_ * HD * HD;
    float* attn = ev + SQ;

    dim3 g1((M_TOTAL + BM - 1) / BM, 1536 / BN);
    qkv_gemm_kernel<<<g1, 256, 0, stream>>>(x, Wqkv, bqkv, qb, kb, vb);

    ksoftmax_stats_kernel<<<(BATCH * HEADS_ * HD) / 256, 256, 0, stream>>>(kb, kmax, ksum);

    context_kernel<<<BATCH * HEADS_, 256, 0, stream>>>(kb, vb, kmax, ksum, ctx);

    crpe_conv_kernel<<<(BATCH * HEADS_ * IMG * IMG * HD) / 256, 256, 0, stream>>>(
        vb, w3, b3, w5, b5, w7, b7, ev);

    dim3 g5(BATCH * HEADS_, (SEQ + 127) / 128);
    factor_crpe_kernel<<<g5, 256, 0, stream>>>(qb, ctx, ev, attn);

    dim3 g6((M_TOTAL + BM - 1) / BM, DIM_ / BN);
    out_gemm_kernel<<<g6, 256, 0, stream>>>(attn, Wout, bout, (float*)d_out);
}
